// CorrBlock_33440615366821
// MI455X (gfx1250) — matrix operand order repaired and checked by reading, not compiled, not run on hardware
//
#include <hip/hip_runtime.h>
#include <stdint.h>

// Problem constants (from reference): B=1, N=8192, D=128, TK=128, KNN=32, RES=3, LEV=3
#define NPTS 8192
#define DDIM 128
#define TKK  128
#define KNNK 32

typedef __attribute__((ext_vector_type(16))) __bf16 v16bf;
typedef __attribute__((ext_vector_type(8)))  float  v8f;
typedef __attribute__((ext_vector_type(4)))  int    v4i;

// gfx1250 async global->LDS path (ASYNCcnt), if this toolchain exposes it.
#if defined(__has_builtin)
#  if __has_builtin(__builtin_amdgcn_global_load_async_to_lds_b128) && \
      __has_builtin(__builtin_amdgcn_s_wait_asynccnt)
#    define HAVE_ASYNC_LDS 1
#  endif
#endif
#ifndef HAVE_ASYNC_LDS
#  define HAVE_ASYNC_LDS 0
#endif

#if HAVE_ASYNC_LDS
typedef __attribute__((address_space(1))) v4i glb_v4i;
typedef __attribute__((address_space(3))) v4i lds_v4i;
#endif

__device__ __forceinline__ unsigned short f2bf(float f) {
    union { float f; unsigned u; } cv; cv.f = f;
    unsigned u = cv.u;
    unsigned r = u + 0x7FFFu + ((u >> 16) & 1u);   // RNE truncate to bf16
    return (unsigned short)(r >> 16);
}
__device__ __forceinline__ float bf2f(unsigned short s) {
    union { unsigned u; float f; } cv; cv.u = ((unsigned)s) << 16;
    return cv.f;
}
__device__ __forceinline__ __bf16 bfbits(unsigned short s) {
    return __builtin_bit_cast(__bf16, s);
}

// 16-byte global->LDS copy: async on gfx1250 (tracked by ASYNCcnt), else sync.
__device__ __forceinline__ void copy16(const unsigned short* gsrc, unsigned short* ldst) {
#if HAVE_ASYNC_LDS
    __builtin_amdgcn_global_load_async_to_lds_b128(
        (glb_v4i*)gsrc, (lds_v4i*)ldst, 0, 0);
#else
    *(uint4*)ldst = *(const uint4*)gsrc;
#endif
}
__device__ __forceinline__ void wait_async_lds() {
#if HAVE_ASYNC_LDS
    __builtin_amdgcn_s_wait_asynccnt(0);
#endif
}

// Load a 16x32 bf16 A fragment from an LDS row (ISA layout: lanes 0-15 hold
// K {0..7,16..23}, lanes 16-31 hold K {8..15,24..31}).
__device__ __forceinline__ v16bf load_afrag(const unsigned short* Arow, int k0, int hi) {
    v16bf a;
    #pragma unroll
    for (int e = 0; e < 8; ++e) {
        a[e]     = bfbits(Arow[k0 + hi * 8 + e]);
        a[8 + e] = bfbits(Arow[k0 + 16 + hi * 8 + e]);
    }
    return a;
}
// Load a 32x16 bf16 B fragment (lane = N col; lanes 0-15 K {0..7,16..23}, 16-31 K {8..15,24..31}).
__device__ __forceinline__ v16bf load_bfrag(const unsigned short* Brow, int k0, int hi) {
    v16bf b;
    #pragma unroll
    for (int e = 0; e < 8; ++e) {
        b[e]     = bfbits(Brow[k0 + hi * 8 + e]);
        b[8 + e] = bfbits(Brow[k0 + 16 + hi * 8 + e]);
    }
    return b;
}

// ---------------------------------------------------------------------------
// Kernel 0: one-time fp32 -> bf16 convert + transpose: f[d][n] -> ft[n][d].
// Removes per-block conversion work from the GEMM and gives both the A and B
// stagings a layout where every 16B is contiguous in global AND LDS.
// ---------------------------------------------------------------------------
__global__ __launch_bounds__(256)
void convert_bf16_t(const float* __restrict__ f1, const float* __restrict__ f2,
                    unsigned short* __restrict__ f1t, unsigned short* __restrict__ f2t) {
    __shared__ unsigned short tile[64][DDIM + 8];
    const int t = threadIdx.x;
    const bool second = blockIdx.x >= (NPTS / 64);
    const int n0 = (second ? blockIdx.x - NPTS / 64 : blockIdx.x) * 64;
    const float* src = second ? f2 : f1;
    unsigned short* dst = second ? f2t : f1t;
    for (int r = 0; r < 32; ++r) {
        int e = t + 256 * r;
        int n = e & 63, d = e >> 6;
        tile[n][d] = f2bf(src[d * NPTS + n0 + n]);   // coalesced along n
    }
    __syncthreads();
    for (int s = 0; s < 4; ++s) {
        int q = t + 256 * s;
        int n = q >> 4, dg = q & 15;
        union { unsigned short s[8]; uint4 v; } pk;
        #pragma unroll
        for (int e = 0; e < 8; ++e) pk.s[e] = tile[n][dg * 8 + e];
        *(uint4*)&dst[(size_t)(n0 + n) * DDIM + dg * 8] = pk.v;
    }
}

// ---------------------------------------------------------------------------
// Kernel 1: corr[i][j] = (1/sqrt(128)) * <f1[:,i], f2[:,j]>, stored bf16.
// 8 waves, 64x128 WG tile, 32x32 per wave (2x2 WMMA sub-tiles).
// B panels are double-buffered: next panel streams into LDS via async
// global->LDS loads while the current one feeds the WMMAs.
// C repacked through LDS so global stores are coalesced b128.
// corr (128 MB bf16) stays resident in the 192 MB L2 for the top-k pass.
// ---------------------------------------------------------------------------
__global__ __launch_bounds__(256)
void corr_gemm_wmma(const unsigned short* __restrict__ f1t,
                    const unsigned short* __restrict__ f2t,
                    unsigned short* __restrict__ corr) {
    __shared__ unsigned short sA[64][DDIM + 8];         // 17.0 KB
    __shared__ unsigned short sB[2][128][DDIM + 8];     // 68.0 KB (dbl-buffered)
    __shared__ unsigned short stage[128][72];           // 18.0 KB
    const int t  = threadIdx.x;
    const int i0 = blockIdx.x * 64;

    // A panel (64 rows x 128 d, bf16): 1024 x 16B transfers.
    for (int s = 0; s < 4; ++s) {
        int q = t + 256 * s;
        int m = q >> 4, dg = q & 15;
        copy16(&f1t[(size_t)(i0 + m) * DDIM + dg * 8], &sA[m][dg * 8]);
    }
    // First B panel.
    for (int s = 0; s < 8; ++s) {
        int q = t + 256 * s;
        int j = q >> 4, dg = q & 15;
        copy16(&f2t[(size_t)j * DDIM + dg * 8], &sB[0][j][dg * 8]);
    }

    const int w = t >> 5, lane = t & 31;                // wave32
    const int mt = w >> 2, jt = w & 3;                  // 2x4 wave grid
    const int lm = lane & 15, hi = lane >> 4;
    const unsigned short* Arow0 = &sA[mt * 32 + lm][0];
    const unsigned short* Arow1 = &sA[mt * 32 + 16 + lm][0];
    const float scale = 0.0883883476483184f;            // 1/sqrt(128)

    for (int jc = 0; jc < NPTS / 128; ++jc) {
        const int cur = jc & 1;
        const int j0 = jc * 128;
        wait_async_lds();                               // my async copies done
        __syncthreads();                                // panel ready; other buf free
        if (jc + 1 < NPTS / 128) {                      // prefetch next panel
            const size_t base = (size_t)(j0 + 128) * DDIM;
            for (int s = 0; s < 8; ++s) {
                int q = t + 256 * s;
                int j = q >> 4, dg = q & 15;
                copy16(&f2t[base + (size_t)j * DDIM + dg * 8], &sB[1 - cur][j][dg * 8]);
            }
        }
        const unsigned short* Brow0 = &sB[cur][jt * 32 + lm][0];
        const unsigned short* Brow1 = &sB[cur][jt * 32 + 16 + lm][0];

        v8f c00 = {}, c01 = {}, c10 = {}, c11 = {};
        #pragma unroll
        for (int k0 = 0; k0 < DDIM; k0 += 32) {
            v16bf a0 = load_afrag(Arow0, k0, hi);
            v16bf a1 = load_afrag(Arow1, k0, hi);
            v16bf b0 = load_bfrag(Brow0, k0, hi);
            v16bf b1 = load_bfrag(Brow1, k0, hi);
            c00 = __builtin_amdgcn_wmma_f32_16x16x32_bf16(false, a0, false, b0, (short)0, c00, false, false);
            c01 = __builtin_amdgcn_wmma_f32_16x16x32_bf16(false, a0, false, b1, (short)0, c01, false, false);
            c10 = __builtin_amdgcn_wmma_f32_16x16x32_bf16(false, a1, false, b0, (short)0, c10, false, false);
            c11 = __builtin_amdgcn_wmma_f32_16x16x32_bf16(false, a1, false, b1, (short)0, c11, false, false);
        }

        // Repack C into LDS: lane's 8 contiguous-M rows of one column -> 16B.
        {
            const v8f* accs[4] = { &c00, &c01, &c10, &c11 };
            #pragma unroll
            for (int ti = 0; ti < 2; ++ti)
                #pragma unroll
                for (int tj = 0; tj < 2; ++tj) {
                    const v8f& c = *accs[ti * 2 + tj];
                    int col = jt * 32 + tj * 16 + lm;
                    int rowb = mt * 32 + ti * 16 + hi * 8;
                    union { unsigned short s[8]; uint4 v; } pk;
                    #pragma unroll
                    for (int r = 0; r < 8; ++r) pk.s[r] = f2bf(c[r] * scale);
                    *(uint4*)&stage[col][rowb] = pk.v;
                }
        }
        __syncthreads();                                // stage visible

        // Coalesced b128 stores: 1024 stores of 8 bf16, row-major.
        #pragma unroll
        for (int s = 0; s < 4; ++s) {
            int q = t + 256 * s;
            int row = q >> 4, cg = q & 15;
            union { unsigned short s[8]; uint4 v; } pk;
            #pragma unroll
            for (int e = 0; e < 8; ++e) pk.s[e] = stage[cg * 8 + e][row];
            *(uint4*)&corr[(size_t)(i0 + row) * NPTS + j0 + cg * 8] = pk.v;
        }
    }
}

// ---------------------------------------------------------------------------
// Kernel 2: exact top-128 per row via 2-pass byte radix rank-select on the
// 16-bit sortable bf16 key. Slot assignment via prefix-scan (deterministic).
// ---------------------------------------------------------------------------
__global__ __launch_bounds__(256)
void topk_rows(const unsigned short* __restrict__ corr,
               float* __restrict__ tcorr, int* __restrict__ tidx) {
    __shared__ unsigned short keys[NPTS];
    __shared__ unsigned int hist[256];
    __shared__ unsigned int scanG[256], scanE[256];
    __shared__ unsigned int sc[4];   // 0:bstar 1:cumAbove 2:T 3:greater
    const int t = threadIdx.x;
    const int i = blockIdx.x;
    const uint4* row4 = (const uint4*)(corr + (size_t)i * NPTS);

    for (int r = 0; r < NPTS / (256 * 8); ++r) {
        int idx = t + 256 * r;
        union { uint4 v; unsigned short s[8]; } in, outk;
        in.v = row4[idx];
        #pragma unroll
        for (int e = 0; e < 8; ++e) {
            unsigned short u = in.s[e];
            outk.s[e] = (unsigned short)(u ^ ((u & 0x8000u) ? 0xFFFFu : 0x8000u));
        }
        *(uint4*)&keys[idx * 8] = outk.v;
    }
    hist[t] = 0;
    __syncthreads();
    for (int r = 0; r < NPTS / 256; ++r)
        atomicAdd(&hist[keys[t + 256 * r] >> 8], 1u);
    __syncthreads();
    if (t == 0) {
        unsigned cum = 0; unsigned bs = 0, ca = 0;
        for (int b = 255; b >= 0; --b) {
            unsigned c = hist[b];
            if (cum + c >= TKK) { bs = (unsigned)b; ca = cum; break; }
            cum += c;
        }
        sc[0] = bs; sc[1] = ca;
    }
    __syncthreads();
    const unsigned bstar = sc[0], cumAbove = sc[1];
    hist[t] = 0;
    __syncthreads();
    for (int r = 0; r < NPTS / 256; ++r) {
        unsigned k = keys[t + 256 * r];
        if ((k >> 8) == bstar) atomicAdd(&hist[k & 0xFFu], 1u);
    }
    __syncthreads();
    if (t == 0) {
        unsigned cum = cumAbove; unsigned ls = 0, g = cumAbove;
        for (int l = 255; l >= 0; --l) {
            unsigned c = hist[l];
            if (cum + c >= TKK) { ls = (unsigned)l; g = cum; break; }
            cum += c;
        }
        sc[2] = (bstar << 8) | ls;   // rank-128 threshold key
        sc[3] = g;                   // count strictly greater than T
    }
    __syncthreads();
    const unsigned T = sc[2], greater = sc[3], needEq = TKK - greater;

    // Deterministic slot assignment: per-thread counts + block prefix scan.
    unsigned cg = 0, ce = 0;
    for (int r = 0; r < NPTS / 256; ++r) {
        unsigned k = keys[t + 256 * r];
        cg += (k > T); ce += (k == T);
    }
    scanG[t] = cg; scanE[t] = ce;
    __syncthreads();
    for (int off = 1; off < 256; off <<= 1) {
        unsigned vg = (t >= off) ? scanG[t - off] : 0u;
        unsigned ve = (t >= off) ? scanE[t - off] : 0u;
        __syncthreads();
        scanG[t] += vg; scanE[t] += ve;
        __syncthreads();
    }
    unsigned gbase = scanG[t] - cg;   // exclusive prefix
    unsigned ebase = scanE[t] - ce;
    for (int r = 0; r < NPTS / 256; ++r) {
        int j = t + 256 * r;
        unsigned k = keys[j];
        int slot = -1;
        if (k > T) slot = (int)(gbase++);
        else if (k == T) { unsigned p = ebase++; if (p < needEq) slot = (int)(greater + p); }
        if (slot >= 0) {
            unsigned short u = (k & 0x8000u) ? (unsigned short)(k ^ 0x8000u)
                                             : (unsigned short)(k ^ 0xFFFFu);
            tcorr[(size_t)i * TKK + slot] = bf2f(u);
            tidx [(size_t)i * TKK + slot] = j;
        }
    }
}

// ---------------------------------------------------------------------------
// Kernel 3: per-point: voxel features (per-bin fixed-order gather), w_v1 MLP,
// bitonic KNN-32 select, deterministic per-point group partial sums.
// ---------------------------------------------------------------------------
__global__ __launch_bounds__(128)
void branch_pre(const float* __restrict__ xyz2, const float* __restrict__ coords,
                const float* __restrict__ tcorr, const int* __restrict__ tidx,
                const float* __restrict__ w_v1, const float* __restrict__ b_v1,
                const float* __restrict__ w_k,  const float* __restrict__ b_k,
                float* __restrict__ x_pre, float* __restrict__ knn_c,
                float* __restrict__ knn_d, float* __restrict__ part) {
    __shared__ float tc[TKK], tx[TKK], ty[TKK], tz[TKK];
    __shared__ float vox[81];
    __shared__ float skey[TKK]; __shared__ int sidx[TKK];
    __shared__ float kc[KNNK], kx[KNNK], ky[KNNK], kz[KNNK];
    __shared__ float xbuf[TKK];
    __shared__ float ps[128], pq[128];
    const int t = threadIdx.x;
    const int n = blockIdx.x;
    const float c0 = coords[n * 3 + 0], c1 = coords[n * 3 + 1], c2 = coords[n * 3 + 2];
    {
        tc[t] = tcorr[(size_t)n * TKK + t];
        int j = tidx[(size_t)n * TKK + t];
        tx[t] = xyz2[j * 3 + 0]; ty[t] = xyz2[j * 3 + 1]; tz[t] = xyz2[j * 3 + 2];
    }
    __syncthreads();

    // Per-bin gather (fixed candidate order -> deterministic sums).
    if (t < 81) {
        int lev = t / 27, rem = t % 27;
        float r = 0.25f * (float)(1 << lev);
        float s = 0.f, cnt = 0.f;
        for (int k = 0; k < TKK; ++k) {
            float d0 = rintf((tx[k] - c0) / r);
            float d1 = rintf((ty[k] - c1) / r);
            float d2 = rintf((tz[k] - c2) / r);
            if (fabsf(d0) <= 1.f && fabsf(d1) <= 1.f && fabsf(d2) <= 1.f) {
                int ci = ((int)d0 + 1) * 9 + ((int)d1 + 1) * 3 + ((int)d2 + 1);
                if (ci == rem) { s += tc[k]; cnt += 1.f; }
            }
        }
        vox[t] = s / fminf(fmaxf(cnt, 1.f), (float)NPTS);
    }
    __syncthreads();

    // x_pre = w_v1(128x81) @ vox + b_v1
    {
        float acc = b_v1[t];
        for (int c = 0; c < 81; ++c) acc += w_v1[t * 81 + c] * vox[c];
        xbuf[t] = acc;
        x_pre[(size_t)t * NPTS + n] = acc;
    }

    // KNN-32: bitonic sort of 128 squared distances (ascending)
    skey[t] = (tx[t]-c0)*(tx[t]-c0) + (ty[t]-c1)*(ty[t]-c1) + (tz[t]-c2)*(tz[t]-c2);
    sidx[t] = t;
    for (int k = 2; k <= TKK; k <<= 1)
        for (int j = k >> 1; j > 0; j >>= 1) {
            __syncthreads();
            int p = t ^ j;
            if (p > t) {
                bool up = ((t & k) == 0);
                float a = skey[t], b = skey[p];
                if (up ? (a > b) : (a < b)) {
                    skey[t] = b; skey[p] = a;
                    int tmp = sidx[t]; sidx[t] = sidx[p]; sidx[p] = tmp;
                }
            }
        }
    __syncthreads();
    if (t < KNNK) {
        int kk = sidx[t];
        kc[t] = tc[kk];
        kx[t] = tx[kk] - c0; ky[t] = ty[kk] - c1; kz[t] = tz[kk] - c2;
        knn_c[(size_t)n * KNNK + t] = kc[t];
        knn_d[(size_t)n * KNNK * 3 + t * 3 + 0] = kx[t];
        knn_d[(size_t)n * KNNK * 3 + t * 3 + 1] = ky[t];
        knn_d[(size_t)n * KNNK * 3 + t * 3 + 2] = kz[t];
    }
    __syncthreads();

    // gn2 partials over y = w_k(64x4)@[corr,dx,dy,dz]+b_k; fixed-order reduce.
    {
        int o = t >> 1, half = t & 1;
        float s = 0.f, q = 0.f;
        for (int k = half * 16; k < half * 16 + 16; ++k) {
            float y = w_k[o*4+0]*kc[k] + w_k[o*4+1]*kx[k] + w_k[o*4+2]*ky[k]
                    + w_k[o*4+3]*kz[k] + b_k[o];
            s += y; q += y * y;
        }
        ps[t] = s; pq[t] = q;
    }
    __syncthreads();
    if (t < 8) {   // group g = threads [16g,16g+16)  (o in [8g,8g+8))
        float s2 = 0.f, q2 = 0.f;
        for (int e = 0; e < 16; ++e) { s2 += ps[t * 16 + e]; q2 += pq[t * 16 + e]; }
        float s1 = 0.f, q1 = 0.f;
        for (int c = 0; c < 16; ++c) { float v = xbuf[t * 16 + c]; s1 += v; q1 += v * v; }
        part[(size_t)n * 32 + t]      = s1;
        part[(size_t)n * 32 + 8 + t]  = q1;
        part[(size_t)n * 32 + 16 + t] = s2;
        part[(size_t)n * 32 + 24 + t] = q2;
    }
}

// ---------------------------------------------------------------------------
// Kernel 4: deterministic fixed-order reduction of per-point group partials.
// ---------------------------------------------------------------------------
__global__ __launch_bounds__(256)
void reduce_stats(const float* __restrict__ part, float* __restrict__ gstats) {
    __shared__ float red[256];
    const int t = threadIdx.x, s = blockIdx.x;
    float acc = 0.f;
    for (int p = t; p < NPTS; p += 256) acc += part[(size_t)p * 32 + s];
    red[t] = acc;
    __syncthreads();
    for (int off = 128; off > 0; off >>= 1) {
        if (t < off) red[t] += red[t + off];
        __syncthreads();
    }
    if (t == 0) gstats[s] = red[0];
}

// ---------------------------------------------------------------------------
// Kernel 5: finalize per point: gn1+PReLU -> w_v2; recompute y, gn2+PReLU,
// max over k -> w_o; out = vox_out + knn_out.
// ---------------------------------------------------------------------------
__global__ __launch_bounds__(128)
void finalize(const float* __restrict__ x_pre, const float* __restrict__ knn_c,
              const float* __restrict__ knn_d, const float* __restrict__ gstats,
              const float* __restrict__ gn1_g, const float* __restrict__ gn1_b,
              const float* __restrict__ p1,
              const float* __restrict__ w_v2, const float* __restrict__ b_v2,
              const float* __restrict__ w_k,  const float* __restrict__ b_k,
              const float* __restrict__ gn2_g, const float* __restrict__ gn2_b,
              const float* __restrict__ p2,
              const float* __restrict__ w_o,  const float* __restrict__ b_o,
              float* __restrict__ out) {
    __shared__ float mean1[8], inv1[8], mean2[8], inv2[8];
    __shared__ float xbuf[128], kc[KNNK], kx[KNNK], ky[KNNK], kz[KNNK];
    __shared__ float pm[128], ym[64];
    const int t = threadIdx.x;
    const int n = blockIdx.x;
    if (t < 8) {
        float cnt = 16.f * (float)NPTS;                 // gn1: 16 ch x N
        float m = gstats[t] / cnt;
        float v = gstats[8 + t] / cnt - m * m;
        mean1[t] = m; inv1[t] = rsqrtf(v + 1e-5f);
    } else if (t < 16) {
        int g = t - 8;
        float cnt = 8.f * (float)NPTS * (float)KNNK;    // gn2: 8 ch x N x K
        float m = gstats[16 + g] / cnt;
        float v = gstats[24 + g] / cnt - m * m;
        mean2[g] = m; inv2[g] = rsqrtf(v + 1e-5f);
    }
    if (t < KNNK) {
        kc[t] = knn_c[(size_t)n * KNNK + t];
        kx[t] = knn_d[(size_t)n * KNNK * 3 + t * 3 + 0];
        ky[t] = knn_d[(size_t)n * KNNK * 3 + t * 3 + 1];
        kz[t] = knn_d[(size_t)n * KNNK * 3 + t * 3 + 2];
    }
    const float a1 = p1[0], a2 = p2[0];
    __syncthreads();
    {
        float x = x_pre[(size_t)t * NPTS + n];
        int g = t >> 4;
        float xn = (x - mean1[g]) * inv1[g] * gn1_g[t] + gn1_b[t];
        xbuf[t] = (xn >= 0.f) ? xn : a1 * xn;
    }
    __syncthreads();
    float voxo = 0.f;
    if (t < 64) {
        voxo = b_v2[t];
        for (int c = 0; c < 128; ++c) voxo += w_v2[t * 128 + c] * xbuf[c];
    }
    {   // two threads per output channel, each covering 16 of the 32 k's
        int o = t >> 1, half = t & 1;
        int g = o >> 3;
        float m = -3.4e38f;
        for (int k = half * 16; k < half * 16 + 16; ++k) {
            float y = w_k[o*4+0]*kc[k] + w_k[o*4+1]*kx[k] + w_k[o*4+2]*ky[k]
                    + w_k[o*4+3]*kz[k] + b_k[o];
            float yn = (y - mean2[g]) * inv2[g] * gn2_g[o] + gn2_b[o];
            yn = (yn >= 0.f) ? yn : a2 * yn;
            m = fmaxf(m, yn);
        }
        pm[t] = m;
    }
    __syncthreads();
    if (t < 64) ym[t] = fmaxf(pm[2 * t], pm[2 * t + 1]);
    __syncthreads();
    if (t < 64) {
        float ko = b_o[t];
        for (int c = 0; c < 64; ++c) ko += w_o[t * 64 + c] * ym[c];
        out[(size_t)t * NPTS + n] = voxo + ko;
    }
}

// ---------------------------------------------------------------------------
extern "C" void kernel_launch(void* const* d_in, const int* in_sizes, int n_in,
                              void* d_out, int out_size, void* d_ws, size_t ws_size,
                              hipStream_t stream) {
    (void)in_sizes; (void)n_in; (void)out_size; (void)ws_size;
    const float* f1     = (const float*)d_in[0];
    const float* f2     = (const float*)d_in[1];
    const float* xyz2   = (const float*)d_in[2];
    const float* coords = (const float*)d_in[3];
    const float* w_v1   = (const float*)d_in[4];
    const float* b_v1   = (const float*)d_in[5];
    const float* gn1_g  = (const float*)d_in[6];
    const float* gn1_b  = (const float*)d_in[7];
    const float* p1     = (const float*)d_in[8];
    const float* w_v2   = (const float*)d_in[9];
    const float* b_v2   = (const float*)d_in[10];
    const float* w_k    = (const float*)d_in[11];
    const float* b_k    = (const float*)d_in[12];
    const float* gn2_g  = (const float*)d_in[13];
    const float* gn2_b  = (const float*)d_in[14];
    const float* p2     = (const float*)d_in[15];
    const float* w_o    = (const float*)d_in[16];
    const float* b_o    = (const float*)d_in[17];
    float* out = (float*)d_out;

    // Workspace layout (≈149 MB total; corr kept bf16 so it fits in 192MB L2)
    char* ws = (char*)d_ws;
    size_t off = 0;
    unsigned short* corr = (unsigned short*)(ws + off); off += (size_t)NPTS * NPTS * 2;
    float* tcorr  = (float*)(ws + off); off += (size_t)NPTS * TKK * 4;
    int*   tidx   = (int*)  (ws + off); off += (size_t)NPTS * TKK * 4;
    float* x_pre  = (float*)(ws + off); off += (size_t)128 * NPTS * 4;
    float* knn_c  = (float*)(ws + off); off += (size_t)NPTS * KNNK * 4;
    float* knn_d  = (float*)(ws + off); off += (size_t)NPTS * KNNK * 3 * 4;
    float* part   = (float*)(ws + off); off += (size_t)NPTS * 32 * 4;
    float* gstats = (float*)(ws + off); off += 128;
    unsigned short* f1t = (unsigned short*)(ws + off); off += (size_t)NPTS * DDIM * 2;
    unsigned short* f2t = (unsigned short*)(ws + off); off += (size_t)NPTS * DDIM * 2;

    convert_bf16_t<<<2 * (NPTS / 64), 256, 0, stream>>>(f1, f2, f1t, f2t);
    corr_gemm_wmma<<<NPTS / 64, 256, 0, stream>>>(f1t, f2t, corr);
    topk_rows<<<NPTS, 256, 0, stream>>>(corr, tcorr, tidx);
    branch_pre<<<NPTS, 128, 0, stream>>>(xyz2, coords, tcorr, tidx,
                                         w_v1, b_v1, w_k, b_k,
                                         x_pre, knn_c, knn_d, part);
    reduce_stats<<<32, 256, 0, stream>>>(part, gstats);
    finalize<<<NPTS, 128, 0, stream>>>(x_pre, knn_c, knn_d, gstats,
                                       gn1_g, gn1_b, p1, w_v2, b_v2,
                                       w_k, b_k, gn2_g, gn2_b, p2,
                                       w_o, b_o, out);
}
